// CausalSelfAttention_4475355922513
// MI455X (gfx1250) — matrix operand order repaired and checked by reading, not compiled, not run on hardware
//
#include <hip/hip_runtime.h>
#include <hip/hip_bf16.h>
#include <cstdint>
#include <cstddef>

// ---------------------------------------------------------------------------
// Problem constants (match reference)
// ---------------------------------------------------------------------------
#define B_    2
#define S_    2048
#define H_    2048
#define NH_   16
#define NKV_  4
#define HD_   128
#define NREP_ (NH_ / NKV_)

static constexpr float NEG_SCORE = -1.0e9f;
static constexpr float QK_SCALE  = 0.08838834764831845f; // 1/sqrt(128)
static constexpr float LOG_BASE  = 9.210340371976184f;   // ln(10000)

typedef __bf16 bf16;
typedef __bf16 v16bf __attribute__((ext_vector_type(16)));
typedef float  v8f   __attribute__((ext_vector_type(8)));

// Output modes for the GEMM epilogue
#define OUT_BF16       0
#define OUT_F32_       1
#define OUT_TRANS_BF16 2

// ---------------------------------------------------------------------------
// WMMA helpers (CDNA5 wave32; this code's contiguous k order, not the ISA table order)
//  A (16x32 bf16): row = lane&15; elem i -> K = khi + i  (khi = 16 if lane>=16)
//  B (32x16 bf16): col = lane&15; elem i -> K = khi + i  (khi = 16 if lane>=16)
//  C/D (16x16 f32): col = lane&15; vgpr i -> row = i + (lane>=16 ? 8 : 0)
// ---------------------------------------------------------------------------
__device__ __forceinline__ v8f vzero() {
  v8f z;
#pragma unroll
  for (int i = 0; i < 8; ++i) z[i] = 0.0f;
  return z;
}

// A fragment from row-major [16][ld] tile (one contiguous 32B read per lane)
__device__ __forceinline__ v16bf ldA(const bf16* __restrict__ p, int ld, int kbase) {
  const int lane = threadIdx.x & 31;
  const int row  = lane & 15;
  const int khi  = (lane & 16) ? 16 : 0;
  const bf16* r = p + (size_t)row * ld + kbase;
  v16bf a;
#pragma unroll
  for (int i = 0; i < 8; ++i) {
    a[i]     = r[khi + i];
    a[8 + i] = r[khi + 8 + i];
  }
  return a;
}

// B fragment where the B matrix is stored TRANSPOSED in LDS, i.e. p is a
// row-major [N][ld] tile holding B^T: element (k, n) = p[n*ld + kbase + k].
// One contiguous 32B read per lane -> 2x ds_load_b128.
__device__ __forceinline__ v16bf ldBT(const bf16* __restrict__ p, int ld, int kbase, int col0) {
  const int lane = threadIdx.x & 31;
  const int n    = col0 + (lane & 15);
  const int khi  = (lane & 16) ? 16 : 0;
  const bf16* r = p + (size_t)n * ld + kbase + khi;
  v16bf b;
#pragma unroll
  for (int i = 0; i < 16; ++i) b[i] = r[i];
  return b;
}

__device__ __forceinline__ v8f wmma_bf16(v16bf a, v16bf b, v8f c) {
  return __builtin_amdgcn_wmma_f32_16x16x32_bf16(
      /*neg_a=*/false, a, /*neg_b=*/false, b,
      /*c_mod=*/(short)0, c, /*reuse_a=*/false, /*reuse_b=*/false);
}

// ---------------------------------------------------------------------------
// fp32 -> bf16 conversion (memory-bound, one pass over inputs)
// ---------------------------------------------------------------------------
__global__ void f32_to_bf16(const float* __restrict__ in, bf16* __restrict__ out, int n) {
  int i = blockIdx.x * blockDim.x + threadIdx.x;
  if (i < n) out[i] = (bf16)in[i];
}

// fp32 [R][C] -> bf16 [C][R] tiled transpose-convert (32x32 LDS tiles).
// Block: (32, 8). Grid: (C/32, R/32). R, C multiples of 32 for all uses.
__global__ __launch_bounds__(256)
void f32_to_bf16_transpose(const float* __restrict__ in, bf16* __restrict__ out,
                           int R, int C) {
  __shared__ float tile[32][33];
  const int c0 = blockIdx.x * 32;
  const int r0 = blockIdx.y * 32;
  const int tx = threadIdx.x;
  const int ty = threadIdx.y;
  for (int i = ty; i < 32; i += 8)
    tile[i][tx] = in[(size_t)(r0 + i) * C + c0 + tx];
  __syncthreads();
  for (int i = ty; i < 32; i += 8)
    out[(size_t)(c0 + i) * R + r0 + tx] = (bf16)tile[tx][i];
}

// ---------------------------------------------------------------------------
// Tiled bf16 WMMA GEMM with pre-transposed B:
//   C[M,N] = A[M,K] * B[K,N],  where Bt = B^T is given row-major [N][K].
// Block: 128 threads (4 waves), tile 64(M) x 128(N), K-step 32.
// Wave w -> (wy = w>>1, wx = w&1) owns a 32x64 sub-tile (2x4 WMMA tiles).
// All fragment loads are contiguous ds_load_b128 pairs.
// M % 64 == 0, N % 128 == 0, K % 32 == 0 for all uses here.
// ---------------------------------------------------------------------------
template <int OUT_MODE>
__global__ __launch_bounds__(128)
void gemm_bt(const bf16* __restrict__ A, const bf16* __restrict__ Bt,
             void* __restrict__ Cp, int M, int N, int K) {
  __shared__ __attribute__((aligned(16))) bf16 As[64 * 32];    // [64 m][32 k]
  __shared__ __attribute__((aligned(16))) bf16 Bts[128 * 32];  // [128 n][32 k]

  const int m0 = blockIdx.x * 64;
  const int n0 = blockIdx.y * 128;
  const int w  = threadIdx.x >> 5;
  const int wy = w >> 1;
  const int wx = w & 1;

  v8f acc[2][4];
#pragma unroll
  for (int mi = 0; mi < 2; ++mi)
#pragma unroll
    for (int ni = 0; ni < 4; ++ni) acc[mi][ni] = vzero();

  for (int k0 = 0; k0 < K; k0 += 32) {
    __syncthreads();
    // A tile: 64x32 bf16 = 256 uint4
    for (int u = threadIdx.x; u < 256; u += 128) {
      int row = u >> 2;
      int c8  = (u & 3) << 3;
      *(uint4*)&As[row * 32 + c8] =
          *(const uint4*)&A[(size_t)(m0 + row) * K + k0 + c8];
    }
    // Bt tile: 128x32 bf16 = 512 uint4 (straight vector copy: B pre-transposed)
    for (int u = threadIdx.x; u < 512; u += 128) {
      int row = u >> 2;
      int c8  = (u & 3) << 3;
      *(uint4*)&Bts[row * 32 + c8] =
          *(const uint4*)&Bt[(size_t)(n0 + row) * K + k0 + c8];
    }
    __syncthreads();

    if (k0 + 32 < K) {
      // hint next K-tile into cache (global_prefetch_b8)
      __builtin_prefetch(&A[(size_t)(m0 + (threadIdx.x & 63)) * K + k0 + 32], 0, 1);
      __builtin_prefetch(&Bt[(size_t)(n0 + threadIdx.x) * K + k0 + 32], 0, 1);
    }

    // hoist all fragments, then issue the 8-WMMA chain
    v16bf af[2], bfr[4];
#pragma unroll
    for (int mi = 0; mi < 2; ++mi)
      af[mi] = ldA(&As[(size_t)(wy * 32 + mi * 16) * 32], 32, 0);
#pragma unroll
    for (int ni = 0; ni < 4; ++ni)
      bfr[ni] = ldBT(Bts, 32, 0, wx * 64 + ni * 16);
#pragma unroll
    for (int mi = 0; mi < 2; ++mi)
#pragma unroll
      for (int ni = 0; ni < 4; ++ni)
        acc[mi][ni] = wmma_bf16(af[mi], bfr[ni], acc[mi][ni]);
  }

  const int lane = threadIdx.x & 31;
  const int col  = lane & 15;
  const int rhi  = (lane & 16) ? 8 : 0;
#pragma unroll
  for (int mi = 0; mi < 2; ++mi)
#pragma unroll
    for (int ni = 0; ni < 4; ++ni)
#pragma unroll
      for (int i = 0; i < 8; ++i) {
        const int gm = m0 + wy * 32 + mi * 16 + i + rhi;
        const int gn = n0 + wx * 64 + ni * 16 + col;
        if (OUT_MODE == OUT_F32_)
          ((float*)Cp)[(size_t)gm * N + gn] = acc[mi][ni][i];
        else if (OUT_MODE == OUT_TRANS_BF16)
          ((bf16*)Cp)[(size_t)gn * M + gm] = (bf16)acc[mi][ni][i];
        else
          ((bf16*)Cp)[(size_t)gm * N + gn] = (bf16)acc[mi][ni][i];
      }
}

// ---------------------------------------------------------------------------
// RoPE (in place on bf16 projections). One block per (b*s, head); 64 threads
// each rotate the pair (i, i+64). Q additionally folds in the 1/sqrt(HD) scale.
// ---------------------------------------------------------------------------
__global__ void rope_kernel(bf16* __restrict__ q, int nheads, float scale) {
  const int bsh = blockIdx.x;
  const int h   = bsh % nheads;
  const int bs  = bsh / nheads;
  const int pos = bs % S_;
  const int i   = threadIdx.x; // 0..63

  const size_t base = (size_t)bs * ((size_t)nheads * HD_) + (size_t)h * HD_;
  const float inv = expf(-(2.0f * (float)i / (float)HD_) * LOG_BASE);
  const float ang = (float)pos * inv;
  const float c = cosf(ang);
  const float s = sinf(ang);

  const float x1 = (float)q[base + i];
  const float x2 = (float)q[base + i + 64];
  q[base + i]      = (bf16)((x1 * c - x2 * s) * scale);
  q[base + i + 64] = (bf16)((x2 * c + x1 * s) * scale);
}

// ---------------------------------------------------------------------------
// Flash attention (causal, GQA). Grid: (S/64, NH, B). 128 threads = 4 waves.
// Wave w owns 16 query rows; 64-key KV tiles stream through LDS.
//  - Q, K tiles row-major [keys][hd]: score B-frags (K^T) are contiguous reads
//  - V arrives pre-transposed from the V-projection GEMM: VsT is [hd][keys],
//    so P*V B-frags are contiguous reads too
//  - online softmax row state per-lane matches the C-layout row mapping
// ---------------------------------------------------------------------------
__global__ __launch_bounds__(128)
void flash_attn(const bf16* __restrict__ Q, const bf16* __restrict__ Kc,
                const bf16* __restrict__ VcT, bf16* __restrict__ O) {
  __shared__ __attribute__((aligned(16))) bf16 Qs[64 * HD_];   // [64 q][128 hd]
  __shared__ __attribute__((aligned(16))) bf16 Ks[64 * HD_];   // [64 k][128 hd]
  __shared__ __attribute__((aligned(16))) bf16 VsT[HD_ * 64];  // [128 hd][64 k]
  __shared__ __attribute__((aligned(16))) bf16 Ps[4][16 * 64]; // per-wave P

  const int qt = blockIdx.x;          // query tile (0..31)
  const int h  = blockIdx.y;          // head
  const int b  = blockIdx.z;          // batch
  const int g  = h / NREP_;           // kv head
  const int q0 = qt * 64;
  const int w    = threadIdx.x >> 5;
  const int lane = threadIdx.x & 31;
  const int col  = lane & 15;
  const int hi   = (lane & 16) ? 1 : 0;

  const size_t row0 = (size_t)b * S_;
  const size_t BScols = (size_t)(B_ * S_);

  // load Q tile (64 x 128 bf16 = 1024 uint4), pre-scaled by QK_SCALE in rope
  for (int u = threadIdx.x; u < 1024; u += 128) {
    int r  = u >> 4;
    int c8 = (u & 15) << 3;
    *(uint4*)&Qs[r * HD_ + c8] =
        *(const uint4*)&Q[(row0 + q0 + r) * (size_t)(NH_ * HD_) + (size_t)h * HD_ + c8];
  }

  v8f o[8];
#pragma unroll
  for (int t = 0; t < 8; ++t) o[t] = vzero();
  float rmax[8], rsum[8];
#pragma unroll
  for (int i = 0; i < 8; ++i) { rmax[i] = -1e30f; rsum[i] = 0.0f; }

  const int ntiles = qt + 1; // causal: only KV tiles up to the diagonal
  for (int j = 0; j < ntiles; ++j) {
    const int k0 = j * 64;
    __syncthreads();
    // K tile: row-major [64 keys][128 hd]
    for (int u = threadIdx.x; u < 1024; u += 128) {
      int r  = u >> 4;
      int c8 = (u & 15) << 3;
      *(uint4*)&Ks[r * HD_ + c8] =
          *(const uint4*)&Kc[(row0 + k0 + r) * (size_t)(NKV_ * HD_) + (size_t)g * HD_ + c8];
    }
    // V tile: transposed [128 hd][64 keys], straight vector copy from VcT
    for (int u = threadIdx.x; u < 1024; u += 128) {
      int d  = u >> 3;
      int c8 = (u & 7) << 3;
      *(uint4*)&VsT[d * 64 + c8] =
          *(const uint4*)&VcT[((size_t)g * HD_ + d) * BScols + row0 + k0 + c8];
    }
    __syncthreads();

    // ---- scores: per hd-chunk load 1 A-frag + 4 B-frags, 4 WMMA ----
    v8f s[4];
#pragma unroll
    for (int kt = 0; kt < 4; ++kt) s[kt] = vzero();
#pragma unroll
    for (int kc = 0; kc < 4; ++kc) {
      v16bf aq = ldA(&Qs[(size_t)(w * 16) * HD_], HD_, kc * 32);
      v16bf bk[4];
#pragma unroll
      for (int kt = 0; kt < 4; ++kt) bk[kt] = ldBT(Ks, HD_, kc * 32, kt * 16);
#pragma unroll
      for (int kt = 0; kt < 4; ++kt) s[kt] = wmma_bf16(aq, bk[kt], s[kt]);
    }

    // ---- causal mask + online softmax ----
    float corr[8];
#pragma unroll
    for (int i = 0; i < 8; ++i) {
      const int qrow = q0 + w * 16 + i + 8 * hi;
      float tm = -1e30f;
#pragma unroll
      for (int kt = 0; kt < 4; ++kt) {
        const int key = k0 + kt * 16 + col;
        float sv = s[kt][i];
        if (key > qrow) sv = NEG_SCORE;
        s[kt][i] = sv;
        tm = fmaxf(tm, sv);
      }
#pragma unroll
      for (int m = 1; m < 16; m <<= 1) tm = fmaxf(tm, __shfl_xor(tm, m, 32));
      const float nm = fmaxf(rmax[i], tm);
      corr[i] = __expf(rmax[i] - nm);
      rmax[i] = nm;
      float ps = 0.0f;
#pragma unroll
      for (int kt = 0; kt < 4; ++kt) {
        const float p = __expf(s[kt][i] - nm);
        s[kt][i] = p;
        ps += p;
      }
#pragma unroll
      for (int m = 1; m < 16; m <<= 1) ps += __shfl_xor(ps, m, 32);
      rsum[i] = rsum[i] * corr[i] + ps;
    }

    // rescale running O accumulators (same row<->element mapping as S)
#pragma unroll
    for (int t = 0; t < 8; ++t)
#pragma unroll
      for (int i = 0; i < 8; ++i) o[t][i] *= corr[i];

    // ---- stage P (16x64 bf16) to wave-private LDS in A-matrix layout ----
    bf16* Pw = Ps[w];
#pragma unroll
    for (int kt = 0; kt < 4; ++kt)
#pragma unroll
      for (int i = 0; i < 8; ++i)
        Pw[(i + 8 * hi) * 64 + kt * 16 + col] = (bf16)s[kt][i];

    // ---- O += P * V : per key-chunk load 1 A-frag + 8 B-frags, 8 WMMA ----
#pragma unroll
    for (int kc = 0; kc < 2; ++kc) {
      v16bf ap = ldA(Pw, 64, kc * 32);
      v16bf bv[8];
#pragma unroll
      for (int t = 0; t < 8; ++t) bv[t] = ldBT(VsT, 64, kc * 32, t * 16);
#pragma unroll
      for (int t = 0; t < 8; ++t) o[t] = wmma_bf16(ap, bv[t], o[t]);
    }
  }

  // ---- normalize and write out (bf16, feeds out-projection GEMM) ----
#pragma unroll
  for (int t = 0; t < 8; ++t)
#pragma unroll
    for (int i = 0; i < 8; ++i) {
      const int r = q0 + w * 16 + i + 8 * hi;
      const size_t dst =
          (row0 + r) * (size_t)(NH_ * HD_) + (size_t)h * HD_ + t * 16 + col;
      O[dst] = (bf16)(o[t][i] / rsum[i]);
    }
}

// ---------------------------------------------------------------------------
// Host launcher
// ---------------------------------------------------------------------------
extern "C" void kernel_launch(void* const* d_in, const int* in_sizes, int n_in,
                              void* d_out, int out_size, void* d_ws, size_t ws_size,
                              hipStream_t stream) {
  (void)in_sizes; (void)n_in; (void)out_size; (void)ws_size;

  const float* x  = (const float*)d_in[0];
  const float* wq = (const float*)d_in[1];
  const float* wk = (const float*)d_in[2];
  const float* wv = (const float*)d_in[3];
  const float* wo = (const float*)d_in[4];

  const int BS = B_ * S_;                 // 4096 rows
  size_t off = 0;
  auto alloc = [&](size_t bytes) -> void* {
    void* p = (char*)d_ws + off;
    off += (bytes + 255) & ~(size_t)255;
    return p;
  };

  bf16* xb   = (bf16*)alloc((size_t)BS * H_ * sizeof(bf16));
  bf16* wqbT = (bf16*)alloc((size_t)(NH_ * HD_) * H_ * sizeof(bf16));  // [2048][2048]
  bf16* wkbT = (bf16*)alloc((size_t)(NKV_ * HD_) * H_ * sizeof(bf16)); // [512][2048]
  bf16* wvbT = (bf16*)alloc((size_t)(NKV_ * HD_) * H_ * sizeof(bf16));
  bf16* wobT = (bf16*)alloc((size_t)H_ * (NH_ * HD_) * sizeof(bf16));  // [2048][2048]
  bf16* Qb   = (bf16*)alloc((size_t)BS * (NH_ * HD_) * sizeof(bf16));  // [4096][2048]
  bf16* Kb   = (bf16*)alloc((size_t)BS * (NKV_ * HD_) * sizeof(bf16)); // [4096][512]
  bf16* VbT  = (bf16*)alloc((size_t)(NKV_ * HD_) * BS * sizeof(bf16)); // [512][4096]
  bf16* Ab   = (bf16*)alloc((size_t)BS * (NH_ * HD_) * sizeof(bf16));  // [4096][2048]

  // x: plain convert (stays row-major as the GEMM A operand)
  {
    size_t n = (size_t)BS * H_;
    f32_to_bf16<<<dim3((unsigned)((n + 255) / 256)), dim3(256), 0, stream>>>(x, xb, (int)n);
  }
  // weights: transpose-convert so the GEMM sees B^T row-major
  {
    dim3 blk(32, 8);
    f32_to_bf16_transpose<<<dim3((NH_ * HD_) / 32, H_ / 32), blk, 0, stream>>>(wq, wqbT, H_, NH_ * HD_);
    f32_to_bf16_transpose<<<dim3((NKV_ * HD_) / 32, H_ / 32), blk, 0, stream>>>(wk, wkbT, H_, NKV_ * HD_);
    f32_to_bf16_transpose<<<dim3((NKV_ * HD_) / 32, H_ / 32), blk, 0, stream>>>(wv, wvbT, H_, NKV_ * HD_);
    f32_to_bf16_transpose<<<dim3(H_ / 32, (NH_ * HD_) / 32), blk, 0, stream>>>(wo, wobT, NH_ * HD_, H_);
  }

  // QKV projections: Q, K row-major bf16; V written transposed for flash
  {
    dim3 gq(BS / 64, (NH_ * HD_) / 128);
    gemm_bt<OUT_BF16><<<gq, 128, 0, stream>>>(xb, wqbT, Qb, BS, NH_ * HD_, H_);
    dim3 gk(BS / 64, (NKV_ * HD_) / 128);
    gemm_bt<OUT_BF16><<<gk, 128, 0, stream>>>(xb, wkbT, Kb, BS, NKV_ * HD_, H_);
    gemm_bt<OUT_TRANS_BF16><<<gk, 128, 0, stream>>>(xb, wvbT, VbT, BS, NKV_ * HD_, H_);
  }

  // RoPE (Q folds in 1/sqrt(HD))
  rope_kernel<<<BS * NH_, 64, 0, stream>>>(Qb, NH_, QK_SCALE);
  rope_kernel<<<BS * NKV_, 64, 0, stream>>>(Kb, NKV_, 1.0f);

  // Flash attention
  {
    dim3 ga(S_ / 64, NH_, B_);
    flash_attn<<<ga, 128, 0, stream>>>(Qb, Kb, VbT, Ab);
  }

  // Output projection (fp32 out -> d_out)
  {
    dim3 go(BS / 64, H_ / 128);
    gemm_bt<OUT_F32_><<<go, 128, 0, stream>>>(Ab, wobT, d_out, BS, H_, NH_ * HD_);
  }
}
